// MovingNCA_14491219657111
// MI455X (gfx1250) — matrix operand order repaired and checked by reading, not compiled, not run on hardware
//
#include <hip/hip_runtime.h>
#include <hip/hip_bf16.h>

typedef __attribute__((ext_vector_type(16))) _Float16 v16h;
typedef __attribute__((ext_vector_type(8)))  _Float16 v8h;
typedef __attribute__((ext_vector_type(8)))  float    v8f;

#define BB        128
#define NN        32
#define N_NEO     30
#define CELLS     900          // 30*30
#define C_STATE   19
#define INPUT_DIM 20
#define IN_FEAT   182          // 9*20+2
#define OUT_DIM   21           // 16+3+2
#define ITERS     10
#define THRESH    0.0007f
#define NTILES    57           // ceil(912/16) rows padded

__device__ __forceinline__ v8f wmma16(v16h a, v16h b, v8f c) {
    return __builtin_amdgcn_wmma_f32_16x16x32_f16(false, a, false, b, (short)0, c, false, false);
}

__global__ __launch_bounds__(256, 1)
void nca_fused_kernel(const float* __restrict__ images,
                      const float* __restrict__ W1, const float* __restrict__ b1,
                      const float* __restrict__ W2, const float* __restrict__ b2,
                      const float* __restrict__ W3, const float* __restrict__ b3,
                      float* __restrict__ out_class, float* __restrict__ out_guess)
{
    // ---- LDS (~187 KB, under 320 KB/WGP on CDNA5) ----
    __shared__ float stateLDS[2][NN * NN * C_STATE];   // 155,648 B (double buffer)
    __shared__ float imgLDS[NN * NN];                  //   4,096 B
    __shared__ int   percX[912];                       //   3,648 B
    __shared__ int   percY[912];                       //   3,648 B
    __shared__ __align__(32) _Float16 w1frag[6 * 2 * 32 * 16]; // 12,288 B
    __shared__ __align__(32) _Float16 w2frag[2 * 32 * 16];     //  2,048 B
    __shared__ __align__(32) _Float16 w3frag[2 * 32 * 16];     //  2,048 B
    __shared__ __align__(32) _Float16 hscratch[8][16 * 32];    //  8,192 B (1 KB/wave)
    __shared__ float b1p[32], b2p[32], b3p[32];

    const int tid  = threadIdx.x;
    const int bat  = blockIdx.x;
    const int lane = tid & 31;
    const int wv   = tid >> 5;
    const int hi   = lane >> 4;       // half-wave id (0/1)
    const int nlo  = lane & 15;

    // ---------------- one-time setup ----------------
    for (int i = tid; i < 2 * NN * NN * C_STATE; i += 256)
        (&stateLDS[0][0])[i] = 0.0f;
    for (int i = tid; i < NN * NN; i += 256)
        imgLDS[i] = images[bat * NN * NN + i];
    for (int i = tid; i < 912; i += 256) {
        int c = (i < CELLS) ? i : 0;
        percX[i] = c / N_NEO;
        percY[i] = c % N_NEO;
    }
    if (tid < 32) {
        b1p[tid] = (tid < 30) ? b1[tid] : 0.0f;
        b2p[tid] = (tid < 30) ? b2[tid] : 0.0f;
        b3p[tid] = (tid < OUT_DIM) ? b3[tid] : 0.0f;
    }
    // Pre-swizzle W1 into WMMA B-operand fragment layout:
    // lane l holds N = 16*t + (l&15); half h holds K = 32*c + (h&7) + 8*(l>>4) + 16*(h>>3)
    for (int e = tid; e < 6 * 2 * 32 * 16; e += 256) {
        int h = e & 15, l = (e >> 4) & 31, t = (e >> 9) & 1, c = e >> 10;
        int K = 32 * c + (h & 7) + 8 * (l >> 4) + 16 * (h >> 3);
        int N = 16 * t + (l & 15);
        float v = (K < IN_FEAT && N < 30) ? W1[K * 30 + N] : 0.0f;
        w1frag[e] = (_Float16)v;
    }
    for (int e = tid; e < 2 * 32 * 16; e += 256) {
        int h = e & 15, l = (e >> 4) & 31, t = (e >> 9) & 1;
        int K = (h & 7) + 8 * (l >> 4) + 16 * (h >> 3);
        int N = 16 * t + (l & 15);
        w2frag[e] = (_Float16)((K < 30 && N < 30) ? W2[K * 30 + N] : 0.0f);
        w3frag[e] = (_Float16)((K < 30 && N < OUT_DIM) ? W3[K * OUT_DIM + N] : 0.0f);
    }
    __syncthreads();

    const v16h* w1f = (const v16h*)w1frag;
    const v16h* w2f = (const v16h*)w2frag;
    const v16h* w3f = (const v16h*)w3frag;
    _Float16* hs = hscratch[wv];

    // ---------------- 10 sequential NCA iterations ----------------
    for (int it = 0; it < ITERS; ++it) {
        const float* cur = stateLDS[it & 1];
        float*       nxt = stateLDS[(it + 1) & 1];
        const bool   last = (it == ITERS - 1);

        for (int tile = wv; tile < NTILES; tile += 8) {
            // ---- gather A (16 rows x 192 K, f16) directly into fragments ----
            const int row  = nlo;
            const int cell = tile * 16 + row;
            const int cc   = (cell < CELLS) ? cell : 0;
            const int gx = cc / N_NEO, gy = cc - (cc / N_NEO) * N_NEO;
            const int px = percX[cc], py = percY[cc];
            const float fposx = (float)(px - 16) * 0.0625f;
            const float fposy = (float)(py - 16) * 0.0625f;

            v16h A[6];
#pragma unroll
            for (int c = 0; c < 6; ++c) {
#pragma unroll
                for (int h = 0; h < 16; ++h) {
                    // ISA 16-bit A layout: K = 32c + h + 8*((h>>3)+(lane>>4))
                    int f = 32 * c + h + 8 * ((h >> 3) + hi);
                    float v;
                    if (f >= IN_FEAT)      v = 0.0f;
                    else if (f == 180)     v = fposx;
                    else if (f == 181)     v = fposy;
                    else {
                        int p  = f / INPUT_DIM;
                        int ch = f - p * INPUT_DIM;
                        int di = p / 3, dj = p - di * 3;
                        v = (ch == 0)
                              ? imgLDS[(px + di) * NN + (py + dj)]
                              : cur[((gx + di) * NN + (gy + dj)) * C_STATE + (ch - 1)];
                    }
                    A[c][h] = (_Float16)v;
                }
            }

            // ---- GEMM1: (16x192) x (192x32) ----
            float bi0 = b1p[nlo], bi1 = b1p[16 + nlo];
            v8f acc0 = {bi0, bi0, bi0, bi0, bi0, bi0, bi0, bi0};
            v8f acc1 = {bi1, bi1, bi1, bi1, bi1, bi1, bi1, bi1};
#pragma unroll
            for (int c = 0; c < 6; ++c) {
                acc0 = wmma16(A[c], w1f[(c * 2 + 0) * 32 + lane], acc0);
                acc1 = wmma16(A[c], w1f[(c * 2 + 1) * 32 + lane], acc1);
            }

            // ---- ReLU, transpose C-layout -> A-layout via per-wave LDS ----
#pragma unroll
            for (int v = 0; v < 8; ++v) {
                int M = v + 8 * hi;
                hs[M * 32 + nlo]      = (_Float16)fmaxf(acc0[v], 0.0f);
                hs[M * 32 + 16 + nlo] = (_Float16)fmaxf(acc1[v], 0.0f);
            }
            asm volatile("s_wait_dscnt 0" ::: "memory"); // DS is in-order per wave; fence compiler too
            union { v16h v; v8h h8[2]; } a2;
            a2.h8[0] = *(const v8h*)&hs[row * 32 + 8 * hi];
            a2.h8[1] = *(const v8h*)&hs[row * 32 + 16 + 8 * hi];
            asm volatile("" ::: "memory");

            // ---- GEMM2: (16x32) x (32x32) ----
            float c20 = b2p[nlo], c21 = b2p[16 + nlo];
            v8f h0 = {c20, c20, c20, c20, c20, c20, c20, c20};
            v8f h1 = {c21, c21, c21, c21, c21, c21, c21, c21};
            h0 = wmma16(a2.v, w2f[0 * 32 + lane], h0);
            h1 = wmma16(a2.v, w2f[1 * 32 + lane], h1);

            asm volatile("" ::: "memory");
#pragma unroll
            for (int v = 0; v < 8; ++v) {
                int M = v + 8 * hi;
                hs[M * 32 + nlo]      = (_Float16)fmaxf(h0[v], 0.0f);
                hs[M * 32 + 16 + nlo] = (_Float16)fmaxf(h1[v], 0.0f);
            }
            asm volatile("s_wait_dscnt 0" ::: "memory");
            union { v16h v; v8h h8[2]; } a3;
            a3.h8[0] = *(const v8h*)&hs[row * 32 + 8 * hi];
            a3.h8[1] = *(const v8h*)&hs[row * 32 + 16 + 8 * hi];
            asm volatile("" ::: "memory");

            // ---- GEMM3: (16x32) x (32x32), valid N < 21 ----
            float c30 = b3p[nlo], c31 = b3p[16 + nlo];
            v8f o0 = {c30, c30, c30, c30, c30, c30, c30, c30};
            v8f o1 = {c31, c31, c31, c31, c31, c31, c31, c31};
            o0 = wmma16(a3.v, w3f[0 * 32 + lane], o0);
            o1 = wmma16(a3.v, w3f[1 * 32 + lane], o1);

            // ---- state / perception / output update (C-layout walk) ----
#pragma unroll
            for (int v = 0; v < 8; ++v) {
                int mcell = tile * 16 + v + 8 * hi;
                if (mcell < CELLS) {
                    int mgx = mcell / N_NEO, mgy = mcell - mgx * N_NEO;
                    int sidx = ((mgx + 1) * NN + (mgy + 1)) * C_STATE;
                    // channel nlo (0..15) from o0
                    nxt[sidx + nlo] = cur[sidx + nlo] + o0[v];
                    int ch1 = 16 + nlo;
                    if (ch1 < C_STATE) {                 // channels 16..18 (classes)
                        float nv = cur[sidx + ch1] + o1[v];
                        nxt[sidx + ch1] = nv;
                        if (last)
                            out_class[((bat * N_NEO + mgx) * N_NEO + mgy) * 3 + (ch1 - 16)] = nv;
                    }
                    if (last) {
                        size_t gbase = (size_t)(bat * CELLS + mcell) * OUT_DIM;
                        out_guess[gbase + nlo] = o0[v];
                        if (ch1 < OUT_DIM) out_guess[gbase + ch1] = o1[v];
                    }
                    if (ch1 == 19) {                     // delta-x channel
                        float g = o1[v];
                        int d = (g > THRESH) ? 1 : ((g < -THRESH) ? -1 : 0);
                        int p = percX[mcell] + d;
                        percX[mcell] = (p < 0) ? 0 : ((p > N_NEO - 1) ? N_NEO - 1 : p);
                    }
                    if (ch1 == 20) {                     // delta-y channel
                        float g = o1[v];
                        int d = (g > THRESH) ? 1 : ((g < -THRESH) ? -1 : 0);
                        int p = percY[mcell] + d;
                        percY[mcell] = (p < 0) ? 0 : ((p > N_NEO - 1) ? N_NEO - 1 : p);
                    }
                }
            }
        }
        __syncthreads();   // state[nxt] complete before next iteration reads it
    }
}

extern "C" void kernel_launch(void* const* d_in, const int* in_sizes, int n_in,
                              void* d_out, int out_size, void* d_ws, size_t ws_size,
                              hipStream_t stream) {
    (void)in_sizes; (void)n_in; (void)out_size; (void)d_ws; (void)ws_size;
    const float* images = (const float*)d_in[0];
    const float* W1 = (const float*)d_in[1];
    const float* b1 = (const float*)d_in[2];
    const float* W2 = (const float*)d_in[3];
    const float* b2 = (const float*)d_in[4];
    const float* W3 = (const float*)d_in[5];
    const float* b3 = (const float*)d_in[6];
    float* out_class = (float*)d_out;
    float* out_guess = out_class + (size_t)BB * N_NEO * N_NEO * 3;   // 345600 floats
    hipLaunchKernelGGL(nca_fused_kernel, dim3(BB), dim3(256), 0, stream,
                       images, W1, b1, W2, b2, W3, b3, out_class, out_guess);
}
